// HistoricalRetrievalModule_10866267259238
// MI455X (gfx1250) — matrix operand order repaired and checked by reading, not compiled, not run on hardware
//
#include <hip/hip_runtime.h>

#define DIM      512
#define BATCH    2048
#define DBN      50000
#define TOPK     5
#define NTILES   3125          // DBN / 16
#define NSPLIT   25
#define TILES_PER_SPLIT 125    // NTILES / NSPLIT
#define WAVES    8
#define BLOCK_T  256
#define MB       128           // rows per sim workgroup (WAVES * 16)
#define BPAD     8             // f16 pad per LDS B row -> conflict-free frag reads
#define BSTRIDE  (DIM + BPAD)  // 520 f16 = 1040 B row stride in LDS
#define CSTRIDE  20            // padded stride for C staging (floats)

typedef __attribute__((ext_vector_type(16))) _Float16 v16h;
typedef __attribute__((ext_vector_type(8)))  _Float16 v8h;
typedef __attribute__((ext_vector_type(8)))  float    v8f;
typedef __attribute__((ext_vector_type(4)))  unsigned int u32x4;
typedef __attribute__((ext_vector_type(8)))  int          i32x8;
typedef __attribute__((ext_vector_type(4)))  int          i32x4;

// ---------------------------------------------------------------------------
// TDM: async-copy one 16-row x 512-f16 tile (row-major, stride 512) from
// global to LDS, inserting 16B padding after each 1024B row so the LDS image
// has row stride BSTRIDE=520 f16.  Tracked by TENSORcnt.
// ---------------------------------------------------------------------------
__device__ __forceinline__ void tdm_load_tile(const _Float16* gsrc, void* lds_dst)
{
  const unsigned long long ga = (unsigned long long)(uintptr_t)gsrc;
  u32x4 g0;
  g0[0] = 1u;                                                  // count=1
  g0[1] = (unsigned)(uintptr_t)lds_dst;                        // LDS byte addr
  g0[2] = (unsigned)(ga & 0xFFFFFFFFu);                        // global[31:0]
  g0[3] = (unsigned)((ga >> 32) & 0x01FFFFFFu) | (2u << 30);   // global[56:32]|type=2

  i32x8 g1;
  g1[0] = (int)((1u << 16)       // data_size = 2B
              | (1u << 20)       // pad_enable
              | (7u << 22)       // pad_interval: 256 DWORDs (=1024B row)
              | (3u << 25));     // pad_amount:   4 DWORDs  (=16B)
  g1[1] = (int)(512u << 16);     // tensor_dim0 = 512
  g1[2] = (int)(16u  << 16);     // tensor_dim1 = 16
  g1[3] = (int)(512u << 16);     // tile_dim0   = 512
  g1[4] = 16;                    // tile_dim1   = 16
  g1[5] = 512;                   // tensor_dim0_stride = 512
  g1[6] = 0;
  g1[7] = 0;

  const i32x4 z4 = {0, 0, 0, 0};
#if __clang_major__ >= 23
  const i32x8 z8 = {0, 0, 0, 0, 0, 0, 0, 0};
  __builtin_amdgcn_tensor_load_to_lds(g0, g1, z4, z4, z8, 0);
#else
  __builtin_amdgcn_tensor_load_to_lds(g0, g1, z4, z4, 0);
#endif
}

// ---------------------------------------------------------------------------
// Kernel 1: row-wise L2 normalize f32 -> f16   (one 256-thread block per row)
// ---------------------------------------------------------------------------
__global__ __launch_bounds__(256) void rownorm_f32_to_f16(
    const float* __restrict__ in, _Float16* __restrict__ out)
{
  const int row = blockIdx.x;
  const float* src = in + (size_t)row * DIM;
  const float v0 = src[threadIdx.x];
  const float v1 = src[threadIdx.x + 256];
  float ss = v0 * v0 + v1 * v1;
  #pragma unroll
  for (int off = 16; off > 0; off >>= 1) ss += __shfl_down(ss, off, 32);
  __shared__ float red[8];
  const int lane = threadIdx.x & 31, wid = threadIdx.x >> 5;
  if (lane == 0) red[wid] = ss;
  __syncthreads();
  float tot = 0.f;
  #pragma unroll
  for (int i = 0; i < 8; ++i) tot += red[i];
  const float inv = 1.0f / fmaxf(sqrtf(tot), 1e-12f);   // x / max(||x||, 1e-12)
  out[(size_t)row * DIM + threadIdx.x]       = (_Float16)(v0 * inv);
  out[(size_t)row * DIM + threadIdx.x + 256] = (_Float16)(v1 * inv);
}

// ---------------------------------------------------------------------------
// Kernel 2: WMMA similarity + fused per-row top-5 (partial per N-split)
//   grid = (BATCH/MB, NSPLIT), block = 256 (8 waves, 16 rows each)
//   TDM double-buffers B tiles in LDS; all 16 B frags preloaded to registers;
//   two independent WMMA accumulation chains.
// ---------------------------------------------------------------------------
__global__ __launch_bounds__(BLOCK_T) void sim_topk_kernel(
    const _Float16* __restrict__ h_norm,
    const _Float16* __restrict__ db_norm,
    float* __restrict__ pvals,
    int*   __restrict__ pidx)
{
  __shared__ _Float16 sB[2][16 * BSTRIDE];       // double-buffered B tiles
  __shared__ float    sC[WAVES][16 * CSTRIDE];   // per-wave C staging for top-k

  const int tid   = threadIdx.x;
  const int wave  = tid >> 5;
  const int lane  = tid & 31;
  const int row16 = lane & 15;
  const int hi    = lane >> 4;
  const int m0    = blockIdx.x * MB;
  const int split = blockIdx.y;
  const int tile0 = split * TILES_PER_SPLIT;

  // Preload all A fragments for this wave's 16 rows (16 frags x 8 VGPRs).
  // A 16x32 f16: lanes 0-15 hold K{k0..k0+7, k0+16..k0+23}; lanes 16-31 rest.
  v16h aF[16];
  {
    const _Float16* abase =
        h_norm + (size_t)(m0 + wave * 16 + row16) * DIM + hi * 8;
    #pragma unroll
    for (int kk = 0; kk < 16; ++kk) {
      union { v16h v; v8h h[2]; } u;
      u.h[0] = *(const v8h*)(abase + kk * 32);
      u.h[1] = *(const v8h*)(abase + kk * 32 + 16);
      aF[kk] = u.v;
    }
  }

  float tv[TOPK]; int ti[TOPK];
  #pragma unroll
  for (int i = 0; i < TOPK; ++i) { tv[i] = -3.0e38f; ti[i] = 0; }

  // Prologue: TDM-copy tile 0 into buffer 0.
  if (wave == 0) {
    tdm_load_tile(db_norm + (size_t)(tile0 * 16) * DIM, &sB[0][0]);
    __builtin_amdgcn_s_wait_tensorcnt(0);
  }
  __syncthreads();

  for (int t = 0; t < TILES_PER_SPLIT; ++t) {
    const int n0 = (tile0 + t) * 16;

    // Kick off async copy of next tile into the other buffer (no wait yet).
    if (wave == 0 && (t + 1) < TILES_PER_SPLIT) {
      tdm_load_tile(db_norm + (size_t)(n0 + 16) * DIM, &sB[(t + 1) & 1][0]);
    }

    const _Float16* bt = &sB[t & 1][0];

    // B 32x16 f16 frag: lane = column; lanes 0-15 hold K{k0..k0+7, k0+16..k0+23};
    // lanes 16-31 hold K{k0+8..k0+15, k0+24..k0+31}.
    // Preload ALL 16 frags: 32 ds_load_b128 issue back-to-back; the MMA chain
    // then needs only partial (in-order) DScnt waits.
    union BF { v16h v; v8h h[2]; };
    BF bf[16];
    #pragma unroll
    for (int kk = 0; kk < 16; ++kk) {
      const _Float16* bb = bt + row16 * BSTRIDE + kk * 32 + hi * 8;
      bf[kk].h[0] = *(const v8h*)(bb);
      bf[kk].h[1] = *(const v8h*)(bb + 16);
    }

    // Two independent accumulation chains (even/odd k) for XDL ILP.
    v8f c0 = {}, c1 = {};
    #pragma unroll
    for (int kk = 0; kk < 16; kk += 2) {
      c0 = __builtin_amdgcn_wmma_f32_16x16x32_f16(
               false, aF[kk],     false, bf[kk].v,     (short)0, c0, false, false);
      c1 = __builtin_amdgcn_wmma_f32_16x16x32_f16(
               false, aF[kk + 1], false, bf[kk + 1].v, (short)0, c1, false, false);
    }
    v8f c;
    #pragma unroll
    for (int j = 0; j < 8; ++j) c[j] = c0[j] + c1[j];

    // Stage C (f32 16x16) to per-wave LDS; lanes 0-15 each own one row's top-5.
    float* st = &sC[wave][0];
    #pragma unroll
    for (int j = 0; j < 8; ++j)               // VGPR j: row j+8*hi, col row16
      st[(j + 8 * hi) * CSTRIDE + row16] = c[j];
    __builtin_amdgcn_wave_barrier();
    asm volatile("" ::: "memory");            // DS same-wave ordering in HW
    if (lane < 16) {
      #pragma unroll
      for (int j = 0; j < 16; ++j) {
        const float v = st[lane * CSTRIDE + j];
        if (v > tv[TOPK - 1]) {
          int p = TOPK - 1;
          while (p > 0 && tv[p - 1] < v) { tv[p] = tv[p-1]; ti[p] = ti[p-1]; --p; }
          tv[p] = v; ti[p] = n0 + j;
        }
      }
    }
    asm volatile("" ::: "memory");
    __builtin_amdgcn_wave_barrier();

    // Next tile's copy must be complete before anyone reads it.
    if (wave == 0) __builtin_amdgcn_s_wait_tensorcnt(0);
    __syncthreads();
  }

  if (lane < 16) {
    const int row = m0 + wave * 16 + lane;
    float* pv = pvals + ((size_t)split * BATCH + row) * TOPK;
    int*   pi = pidx  + ((size_t)split * BATCH + row) * TOPK;
    #pragma unroll
    for (int i = 0; i < TOPK; ++i) { pv[i] = tv[i]; pi[i] = ti[i]; }
  }
}

// ---------------------------------------------------------------------------
// Kernel 3: merge split-partials -> global top-5, softmax, gather, blend
//   grid = BATCH/16, block = 256
// ---------------------------------------------------------------------------
__global__ __launch_bounds__(256) void merge_gather_kernel(
    const float* __restrict__ h_current,
    const float* __restrict__ history_db,
    const float* __restrict__ alpha_p,
    const float* __restrict__ pvals,
    const int*   __restrict__ pidx,
    float* __restrict__ out)
{
  __shared__ float s_mu[16][TOPK];
  __shared__ int   s_ix[16][TOPK];
  const int tid = threadIdx.x;
  const int m0  = blockIdx.x * 16;

  if (tid < 16) {
    const int row = m0 + tid;
    float bv[TOPK]; int bi[TOPK];
    #pragma unroll
    for (int i = 0; i < TOPK; ++i) { bv[i] = -3.0e38f; bi[i] = 0; }
    for (int s = 0; s < NSPLIT; ++s) {
      const float* pv = pvals + ((size_t)s * BATCH + row) * TOPK;
      const int*   pi = pidx  + ((size_t)s * BATCH + row) * TOPK;
      #pragma unroll
      for (int i = 0; i < TOPK; ++i) {
        const float v = pv[i];
        if (v > bv[TOPK - 1]) {
          int p = TOPK - 1;
          while (p > 0 && bv[p - 1] < v) { bv[p] = bv[p-1]; bi[p] = bi[p-1]; --p; }
          bv[p] = v; bi[p] = pi[i];
        } else break;  // partial lists are sorted descending
      }
    }
    const float mx = bv[0];
    float e[TOPK], sum = 0.f;
    #pragma unroll
    for (int i = 0; i < TOPK; ++i) { e[i] = __expf(bv[i] - mx); sum += e[i]; }
    const float inv = 1.f / sum;
    #pragma unroll
    for (int i = 0; i < TOPK; ++i) { s_mu[tid][i] = e[i] * inv; s_ix[tid][i] = bi[i]; }
  }
  __syncthreads();

  const float a = 1.f / (1.f + __expf(-alpha_p[0]));
  for (int e2 = tid; e2 < 16 * DIM; e2 += 256) {
    const int r = e2 >> 9;
    const int d = e2 & (DIM - 1);
    float hr = 0.f;
    #pragma unroll
    for (int i = 0; i < TOPK; ++i)
      hr += s_mu[r][i] * history_db[(size_t)s_ix[r][i] * DIM + d];
    const size_t o = (size_t)(m0 + r) * DIM + d;
    out[o] = a * h_current[o] + (1.f - a) * hr;
  }
}

// ---------------------------------------------------------------------------
extern "C" void kernel_launch(void* const* d_in, const int* in_sizes, int n_in,
                              void* d_out, int out_size, void* d_ws, size_t ws_size,
                              hipStream_t stream)
{
  (void)in_sizes; (void)n_in; (void)out_size; (void)ws_size;
  const float* h_current  = (const float*)d_in[0];
  const float* history_db = (const float*)d_in[1];
  const float* alpha      = (const float*)d_in[2];

  // workspace layout (~55 MB): db_norm f16 | h_norm f16 | pvals f32 | pidx i32
  char* ws = (char*)d_ws;
  _Float16* db_norm = (_Float16*)ws;  ws += (size_t)DBN   * DIM * sizeof(_Float16);
  _Float16* h_norm  = (_Float16*)ws;  ws += (size_t)BATCH * DIM * sizeof(_Float16);
  float*    pvals   = (float*)ws;     ws += (size_t)NSPLIT * BATCH * TOPK * sizeof(float);
  int*      pidx    = (int*)ws;

  rownorm_f32_to_f16<<<DBN,   256, 0, stream>>>(history_db, db_norm);
  rownorm_f32_to_f16<<<BATCH, 256, 0, stream>>>(h_current,  h_norm);

  dim3 grid(BATCH / MB, NSPLIT);
  sim_topk_kernel<<<grid, BLOCK_T, 0, stream>>>(h_norm, db_norm, pvals, pidx);

  merge_gather_kernel<<<BATCH / 16, 256, 0, stream>>>(
      h_current, history_db, alpha, pvals, pidx, (float*)d_out);
}
